// EncoderLayer_40724879900968
// MI455X (gfx1250) — matrix operand order repaired and checked by reading, not compiled, not run on hardware
//
#include <hip/hip_runtime.h>
#include <hip/hip_bf16.h>
#include <math.h>

// ---------------------------------------------------------------- types
typedef __attribute__((ext_vector_type(16))) _Float16 v16h;
typedef __attribute__((ext_vector_type(8)))  _Float16 v8h;
typedef __attribute__((ext_vector_type(8)))  float    v8f;

union AB16 { v16h v; v8h h[2]; };

__device__ inline v8f wmma16(v16h a, v16h b, v8f c) {
    return __builtin_amdgcn_wmma_f32_16x16x32_f16(
        /*neg_a=*/false, a, /*neg_b=*/false, b,
        /*c_mod=*/(short)0, c, /*reuse_a=*/false, /*reuse_b=*/false);
}

__device__ inline v8f v8f_zero() {
    v8f z = {0.f, 0.f, 0.f, 0.f, 0.f, 0.f, 0.f, 0.f};
    return z;
}

#define D_MODEL 1024
#define SEQ     2048
#define NBATCH  2
#define NHEADS  16
#define DHEAD   64
#define DFF     512
#define MROWS   (NBATCH * SEQ)   // 4096

// GEMM register tiling: each wave owns MT x NT tiles of 16x16
#define MT 4
#define NT 4
// 8 waves arranged 4 (M) x 2 (N): block tile = 256 x 128

// ---------------------------------------------------------------- weight transpose+convert
// in: K x N row-major f32  ->  out: N x K row-major f16
__global__ __launch_bounds__(256)
void transpose_f32_to_f16(const float* __restrict__ in, _Float16* __restrict__ out,
                          int K, int N) {
    int n = blockIdx.x * 32 + (threadIdx.x & 31);
    int k = blockIdx.y * 8 + (threadIdx.x >> 5);
    if (n < N && k < K)
        out[(size_t)n * K + k] = (_Float16)in[(size_t)k * N + n];
}

// ---------------------------------------------------------------- layernorm (unbiased var)
__global__ __launch_bounds__(256)
void layernorm_k(const float* __restrict__ x, const float* __restrict__ alpha,
                 const float* __restrict__ beta, float* __restrict__ outF,
                 _Float16* __restrict__ outH) {
    __shared__ float red[256];
    const int row = blockIdx.x;
    const int tid = threadIdx.x;
    const float* xr = x + (size_t)row * D_MODEL;

    float s = 0.f;
    for (int c = tid; c < D_MODEL; c += 256) s += xr[c];
    red[tid] = s; __syncthreads();
    for (int o = 128; o > 0; o >>= 1) { if (tid < o) red[tid] += red[tid + o]; __syncthreads(); }
    float mean = red[0] / (float)D_MODEL;
    __syncthreads();

    float s2 = 0.f;
    for (int c = tid; c < D_MODEL; c += 256) { float d = xr[c] - mean; s2 += d * d; }
    red[tid] = s2; __syncthreads();
    for (int o = 128; o > 0; o >>= 1) { if (tid < o) red[tid] += red[tid + o]; __syncthreads(); }
    float inv = rsqrtf(red[0] / (float)(D_MODEL - 1) + 1e-6f);

    for (int c = tid; c < D_MODEL; c += 256) {
        float v = alpha[c] * (xr[c] - mean) * inv + beta[c];
        size_t idx = (size_t)row * D_MODEL + c;
        if (outF) outF[idx] = v;
        if (outH) outH[idx] = (_Float16)v;
    }
}

// ---------------------------------------------------------------- WMMA GEMM
// C[M,N] = A[M,K](f16) * Bt[N,K]^T(f16) + bias[N]  (+relu) (+resid[M,N])
// 8 waves / block arranged 4x2; wave = MTxNT tiles of 16x16 (64x64);
// block tile 256x128. 16 WMMAs per 16 b128 loads per K-step.
__global__ __launch_bounds__(256)
void gemm_f16_wmma(const _Float16* __restrict__ A, const _Float16* __restrict__ Bt,
                   const float* __restrict__ bias, const float* __restrict__ resid,
                   float* __restrict__ outF, _Float16* __restrict__ outH,
                   int M, int N, int K, int relu) {
    const int lane = threadIdx.x & 31;
    const int wave = threadIdx.x >> 5;   // 0..7
    const int wm = wave >> 1;            // 0..3
    const int wn = wave & 1;             // 0..1
    const int bm = blockIdx.x * (4 * MT * 16) + wm * (MT * 16);
    const int bn = blockIdx.y * (2 * NT * 16) + wn * (NT * 16);
    const int l15 = lane & 15;
    const bool hi = lane >= 16;

    v8f acc[MT][NT];
    for (int i = 0; i < MT; ++i)
        for (int j = 0; j < NT; ++j) acc[i][j] = v8f_zero();

    for (int k0 = 0; k0 < K; k0 += 32) {
        AB16 a[MT], b[NT];
        const int ka = k0 + (hi ? 8 : 0);
        for (int i = 0; i < MT; ++i) {
            const _Float16* ap = A + (size_t)(bm + i * 16 + l15) * K + ka;
            a[i].h[0] = *(const v8h*)(ap);
            a[i].h[1] = *(const v8h*)(ap + 16);
        }
        const int kb = k0 + (hi ? 8 : 0);
        for (int j = 0; j < NT; ++j) {
            const _Float16* bp = Bt + (size_t)(bn + j * 16 + l15) * K + kb;
            b[j].h[0] = *(const v8h*)(bp);
            b[j].h[1] = *(const v8h*)(bp + 16);
        }
        for (int i = 0; i < MT; ++i)
            for (int j = 0; j < NT; ++j)
                acc[i][j] = wmma16(a[i].v, b[j].v, acc[i][j]);
    }

    for (int i = 0; i < MT; ++i)
        for (int j = 0; j < NT; ++j) {
            const int col = bn + j * 16 + l15;
            const float bsv = bias ? bias[col] : 0.f;
            for (int r = 0; r < 8; ++r) {
                const int row = bm + i * 16 + r + (hi ? 8 : 0);
                float v = acc[i][j][r] + bsv;
                if (relu) v = v > 0.f ? v : 0.f;
                const size_t idx = (size_t)row * N + col;
                if (resid) v += resid[idx];
                if (outF) outF[idx] = v;
                if (outH) outH[idx] = (_Float16)v;
            }
        }
}

// ---------------------------------------------------------------- flash attention
// One block: 64 queries of one (b,h). Stream 64-key blocks; online softmax.
// Q/K/V/O layout: [b][s][h*64+d]  (f16, row stride D_MODEL)
__global__ __launch_bounds__(256)
void attn_wmma(const _Float16* __restrict__ Qg, const _Float16* __restrict__ Kg,
               const _Float16* __restrict__ Vg, _Float16* __restrict__ Og) {
    __shared__ __attribute__((aligned(16))) _Float16 Qs[64 * 64];
    __shared__ __attribute__((aligned(16))) _Float16 Vt[64 * 64];  // [d][key]
    __shared__ __attribute__((aligned(16))) float    Sc[64 * 64];
    __shared__ __attribute__((aligned(16))) _Float16 Ps[64 * 64];
    __shared__ float mrow[64], lrow[64], arow[64];

    const int tid  = threadIdx.x;
    const int lane = tid & 31;
    const int wave = tid >> 5;
    const int l15  = lane & 15;
    const bool hi  = lane >= 16;
    const int tm   = wave >> 1;   // row tile 0..3 (16 queries each)
    const int wn   = wave & 1;    // col half (32 of dh)
    const int bh = blockIdx.y;
    const int b = bh >> 4, h = bh & 15;
    const int q0 = blockIdx.x * 64;
    const size_t base = ((size_t)b * SEQ) * D_MODEL + (size_t)h * DHEAD;

    // stage 64x64 Q block into LDS
    {
        const int r = tid >> 2, d0 = (tid & 3) * 16;
        const _Float16* src = Qg + base + (size_t)(q0 + r) * D_MODEL + d0;
        *(v8h*)(Qs + r * 64 + d0)     = *(const v8h*)(src);
        *(v8h*)(Qs + r * 64 + d0 + 8) = *(const v8h*)(src + 8);
    }
    if (tid < 64) { mrow[tid] = -3.0e38f; lrow[tid] = 0.f; }

    v8f o[2];
    o[0] = v8f_zero(); o[1] = v8f_zero();

    for (int j = 0; j < SEQ / 64; ++j) {
        __syncthreads();
        // stage transposed V block: Vt[d][key]
        {
            const int kk = tid >> 2, d0 = (tid & 3) * 16;
            const _Float16* src = Vg + base + (size_t)(j * 64 + kk) * D_MODEL + d0;
            for (int i = 0; i < 16; ++i) Vt[(d0 + i) * 64 + kk] = src[i];
        }
        __syncthreads();

        // S = (Q Kᵀ) / 8 for this wave's two 16x16 tiles
        for (int tt = 0; tt < 2; ++tt) {
            const int cb = wn * 32 + tt * 16;
            v8f s = v8f_zero();
            for (int k0 = 0; k0 < DHEAD; k0 += 32) {
                AB16 a, bf;
                const int ka = k0 + (hi ? 8 : 0);
                const _Float16* ap = Qs + (tm * 16 + l15) * 64 + ka;
                a.h[0] = *(const v8h*)(ap);
                a.h[1] = *(const v8h*)(ap + 16);
                const int kb = k0 + (hi ? 8 : 0);
                const _Float16* bp = Kg + base + (size_t)(j * 64 + cb + l15) * D_MODEL + kb;
                bf.h[0] = *(const v8h*)(bp);
                bf.h[1] = *(const v8h*)(bp + 16);
                s = wmma16(a.v, bf.v, s);
            }
            for (int r = 0; r < 8; ++r)
                Sc[(tm * 16 + r + (hi ? 8 : 0)) * 64 + cb + l15] = s[r] * 0.125f;
        }
        __syncthreads();

        // online softmax update, one row per thread (threads 0..63)
        if (tid < 64) {
            const float* sr = Sc + tid * 64;
            float mx = mrow[tid];
            for (int c = 0; c < 64; ++c) mx = fmaxf(mx, sr[c]);
            const float al = __expf(mrow[tid] - mx);
            float sum = 0.f;
            _Float16* pr = Ps + tid * 64;
            for (int c = 0; c < 64; ++c) {
                const float p = __expf(sr[c] - mx);
                pr[c] = (_Float16)p;
                sum += p;
            }
            lrow[tid] = lrow[tid] * al + sum;
            mrow[tid] = mx;
            arow[tid] = al;
        }
        __syncthreads();

        // O = O*alpha + P @ V
        for (int tt = 0; tt < 2; ++tt) {
            const int cb = wn * 32 + tt * 16;
            for (int r = 0; r < 8; ++r)
                o[tt][r] *= arow[tm * 16 + r + (hi ? 8 : 0)];
            for (int k0 = 0; k0 < 64; k0 += 32) {
                AB16 a, bf;
                const int ka = k0 + (hi ? 8 : 0);
                const _Float16* ap = Ps + (tm * 16 + l15) * 64 + ka;
                a.h[0] = *(const v8h*)(ap);
                a.h[1] = *(const v8h*)(ap + 16);
                const int kb = k0 + (hi ? 8 : 0);
                const _Float16* bp = Vt + (cb + l15) * 64 + kb;
                bf.h[0] = *(const v8h*)(bp);
                bf.h[1] = *(const v8h*)(bp + 16);
                o[tt] = wmma16(a.v, bf.v, o[tt]);
            }
        }
    }
    __syncthreads();

    for (int tt = 0; tt < 2; ++tt) {
        const int cb = wn * 32 + tt * 16;
        for (int r = 0; r < 8; ++r) {
            const int row = tm * 16 + r + (hi ? 8 : 0);
            const float v = o[tt][r] / lrow[row];
            Og[base + (size_t)(q0 + row) * D_MODEL + cb + l15] = (_Float16)v;
        }
    }
}

// ---------------------------------------------------------------- host side
extern "C" void kernel_launch(void* const* d_in, const int* in_sizes, int n_in,
                              void* d_out, int out_size, void* d_ws, size_t ws_size,
                              hipStream_t stream) {
    const float* x      = (const float*)d_in[0];
    const float* Wq     = (const float*)d_in[1];
    const float* bq     = (const float*)d_in[2];
    const float* Wk     = (const float*)d_in[3];
    const float* bk     = (const float*)d_in[4];
    const float* Wv     = (const float*)d_in[5];
    const float* bv     = (const float*)d_in[6];
    const float* Wo     = (const float*)d_in[7];
    const float* bo     = (const float*)d_in[8];
    const float* alpha1 = (const float*)d_in[9];
    const float* beta1  = (const float*)d_in[10];
    const float* alpha2 = (const float*)d_in[11];
    const float* beta2  = (const float*)d_in[12];
    const float* W1     = (const float*)d_in[13];
    const float* b1     = (const float*)d_in[14];
    const float* W2     = (const float*)d_in[15];
    const float* b2     = (const float*)d_in[16];

    char* w = (char*)d_ws;
    auto alloc = [&](size_t bytes) -> void* {
        void* p = (void*)w;
        w += (bytes + 255) & ~(size_t)255;
        return p;
    };
    const size_t MD = (size_t)MROWS * D_MODEL;

    float*    xn_f  = (float*)alloc(MD * 4);            // norm1(x) f32 (residual)
    float*    x1_f  = (float*)alloc(MD * 4);            // after attn residual
    _Float16* xn_h  = (_Float16*)alloc(MD * 2);
    _Float16* q_h   = (_Float16*)alloc(MD * 2);
    _Float16* k_h   = (_Float16*)alloc(MD * 2);
    _Float16* v_h   = (_Float16*)alloc(MD * 2);
    _Float16* ao_h  = (_Float16*)alloc(MD * 2);         // attention output
    _Float16* n2_h  = (_Float16*)alloc(MD * 2);         // norm2 output
    _Float16* g_h   = (_Float16*)alloc((size_t)MROWS * DFF * 2);  // relu(h W1 + b1)
    _Float16* Wq_t  = (_Float16*)alloc((size_t)D_MODEL * D_MODEL * 2);
    _Float16* Wk_t  = (_Float16*)alloc((size_t)D_MODEL * D_MODEL * 2);
    _Float16* Wv_t  = (_Float16*)alloc((size_t)D_MODEL * D_MODEL * 2);
    _Float16* Wo_t  = (_Float16*)alloc((size_t)D_MODEL * D_MODEL * 2);
    _Float16* W1_t  = (_Float16*)alloc((size_t)DFF * D_MODEL * 2);
    _Float16* W2_t  = (_Float16*)alloc((size_t)D_MODEL * DFF * 2);

    // weight transpose + f16 convert
    transpose_f32_to_f16<<<dim3(D_MODEL / 32, D_MODEL / 8), 256, 0, stream>>>(Wq, Wq_t, D_MODEL, D_MODEL);
    transpose_f32_to_f16<<<dim3(D_MODEL / 32, D_MODEL / 8), 256, 0, stream>>>(Wk, Wk_t, D_MODEL, D_MODEL);
    transpose_f32_to_f16<<<dim3(D_MODEL / 32, D_MODEL / 8), 256, 0, stream>>>(Wv, Wv_t, D_MODEL, D_MODEL);
    transpose_f32_to_f16<<<dim3(D_MODEL / 32, D_MODEL / 8), 256, 0, stream>>>(Wo, Wo_t, D_MODEL, D_MODEL);
    transpose_f32_to_f16<<<dim3(DFF / 32, D_MODEL / 8), 256, 0, stream>>>(W1, W1_t, D_MODEL, DFF);
    transpose_f32_to_f16<<<dim3(D_MODEL / 32, DFF / 8), 256, 0, stream>>>(W2, W2_t, DFF, D_MODEL);

    // norm1 -> xn (f32 + f16)
    layernorm_k<<<MROWS, 256, 0, stream>>>(x, alpha1, beta1, xn_f, xn_h);

    // Q, K, V projections: block tile 256x128
    dim3 gQKV(MROWS / 256, D_MODEL / 128);
    gemm_f16_wmma<<<gQKV, 256, 0, stream>>>(xn_h, Wq_t, bq, nullptr, nullptr, q_h,
                                            MROWS, D_MODEL, D_MODEL, 0);
    gemm_f16_wmma<<<gQKV, 256, 0, stream>>>(xn_h, Wk_t, bk, nullptr, nullptr, k_h,
                                            MROWS, D_MODEL, D_MODEL, 0);
    gemm_f16_wmma<<<gQKV, 256, 0, stream>>>(xn_h, Wv_t, bv, nullptr, nullptr, v_h,
                                            MROWS, D_MODEL, D_MODEL, 0);

    // flash attention
    attn_wmma<<<dim3(SEQ / 64, NBATCH * NHEADS), 256, 0, stream>>>(q_h, k_h, v_h, ao_h);

    // Wo projection + residual -> x1 (f32)
    gemm_f16_wmma<<<gQKV, 256, 0, stream>>>(ao_h, Wo_t, bo, xn_f, x1_f, nullptr,
                                            MROWS, D_MODEL, D_MODEL, 0);

    // norm2 -> f16
    layernorm_k<<<MROWS, 256, 0, stream>>>(x1_f, alpha2, beta2, nullptr, n2_h);

    // FFN
    gemm_f16_wmma<<<dim3(MROWS / 256, DFF / 128), 256, 0, stream>>>(
        n2_h, W1_t, b1, nullptr, nullptr, g_h, MROWS, DFF, D_MODEL, 1);
    gemm_f16_wmma<<<gQKV, 256, 0, stream>>>(g_h, W2_t, b2, x1_f, (float*)d_out, nullptr,
                                            MROWS, D_MODEL, DFF, 0);
}
